// Attention_3427383902491
// MI455X (gfx1250) — matrix operand order repaired and checked by reading, not compiled, not run on hardware
//
#include <hip/hip_runtime.h>

// ---------------------------------------------------------------------------
// Problem constants (match reference: B=2, L=2048, H=2048, NH=16, HD=128)
// ---------------------------------------------------------------------------
#define BATCH   2
#define SEQ     2048
#define HID     2048
#define NHEADS  16
#define HDIM    128
#define QKVDIM  (3 * HID)        // 6144
#define ROWS    (BATCH * SEQ)    // 4096

typedef __bf16 bf16_t;
typedef __attribute__((ext_vector_type(16))) __bf16 v16bf;
typedef __attribute__((ext_vector_type(8)))  __bf16 v8bf;
typedef __attribute__((ext_vector_type(4)))  __bf16 v4bf;
typedef __attribute__((ext_vector_type(8)))  float  v8f;
typedef __attribute__((ext_vector_type(4)))  unsigned int tdm_g0_t;
typedef __attribute__((ext_vector_type(8)))  unsigned int tdm_g1_t;

// ---------------------------------------------------------------------------
// bf16 helpers (bit-level, round-to-nearest-even)
// ---------------------------------------------------------------------------
static __device__ __forceinline__ float bf2f(bf16_t x) {
    unsigned short u = __builtin_bit_cast(unsigned short, x);
    unsigned int v = ((unsigned int)u) << 16;
    return __builtin_bit_cast(float, v);
}
static __device__ __forceinline__ bf16_t f2bf(float f) {
    unsigned int u = __builtin_bit_cast(unsigned int, f);
    unsigned int r = u + 0x7FFFu + ((u >> 16) & 1u);
    unsigned short s = (unsigned short)(r >> 16);
    return __builtin_bit_cast(bf16_t, s);
}

static __device__ __forceinline__ v16bf cat16(v8bf lo, v8bf hi) {
    v16bf r;
#pragma unroll
    for (int i = 0; i < 8; ++i) { r[i] = lo[i]; r[i + 8] = hi[i]; }
    return r;
}
// A-fragment (16x32 bf16): per lane, 16 contiguous halfs
static __device__ __forceinline__ v16bf load_a_frag(const bf16_t* p) {
    v8bf lo = *reinterpret_cast<const v8bf*>(p);
    v8bf hi = *reinterpret_cast<const v8bf*>(p + 8);
    return cat16(lo, hi);
}
// B-fragment (32x16 bf16): per lane, 16 contiguous halfs
static __device__ __forceinline__ v16bf load_b_frag(const bf16_t* p) {
    v8bf lo = *reinterpret_cast<const v8bf*>(p);
    v8bf hi = *reinterpret_cast<const v8bf*>(p + 8);
    return cat16(lo, hi);
}
static __device__ __forceinline__ v8f zero8() {
    v8f z;
#pragma unroll
    for (int i = 0; i < 8; ++i) z[i] = 0.0f;
    return z;
}

// ---------------------------------------------------------------------------
// Tensor Data Mover: issue a 2D tile load (rows x tile_k bf16) into LDS.
// D# per CDNA5 ISA ch.8: group0 = {count, lds_addr, global_addr, type=2},
// group1 = {data_size=1(2B), tensor dims, tile dims, dim0 stride}.
// Rows land packed contiguously in LDS (row stride = tile_k elements).
// ---------------------------------------------------------------------------
static __device__ __forceinline__ void tdm_load_tile_2d(
    unsigned lds_off_bytes, const bf16_t* gaddr,
    unsigned k_elems, unsigned rows_total,
    unsigned tile_k, unsigned tile_rows)
{
    unsigned long long ga = (unsigned long long)gaddr;
    tdm_g0_t g0;
    g0[0] = 1u;                                             // count=1, user mode
    g0[1] = lds_off_bytes;                                  // lds_addr
    g0[2] = (unsigned)ga;                                   // global_addr[31:0]
    g0[3] = (unsigned)((ga >> 32) & 0x01FFFFFFu) | 0x80000000u; // addr[56:32] | type=2
    tdm_g1_t g1;
    g1[0] = 0x00010000u;                                    // data_size=1 (2 bytes)
    g1[1] = (k_elems & 0xFFFFu) << 16;                      // tensor_dim0[15:0]
    g1[2] = (k_elems >> 16) | ((rows_total & 0xFFFFu) << 16); // dim0[31:16] | dim1[15:0]
    g1[3] = (rows_total >> 16) | (tile_k << 16);            // dim1[31:16] | tile_dim0
    g1[4] = tile_rows;                                      // tile_dim1 (tile_dim2=0)
    g1[5] = k_elems;                                        // tensor_dim0_stride[31:0]
    g1[6] = 0u;                                             // stride hi / dim1_stride lo
    g1[7] = 0u;
    asm volatile("tensor_load_to_lds %0, %1"
                 :: "s"(g0), "s"(g1)
                 : "memory");
}

// ---------------------------------------------------------------------------
// Kernel 1: fp32 -> bf16 conversion (vectorized x4)
// ---------------------------------------------------------------------------
__global__ void cvt_f32_to_bf16(const float* __restrict__ src,
                                bf16_t* __restrict__ dst, long n4) {
    long i = (long)blockIdx.x * blockDim.x + threadIdx.x;
    if (i >= n4) return;
    float4 f = reinterpret_cast<const float4*>(src)[i];
    v4bf o;
    o[0] = f2bf(f.x); o[1] = f2bf(f.y); o[2] = f2bf(f.z); o[3] = f2bf(f.w);
    reinterpret_cast<v4bf*>(dst)[i] = o;
}

// ---------------------------------------------------------------------------
// Kernel 2: LDS-tiled WMMA GEMM with TDM double-buffered staging.
//   C[M,N] = A[M,K] * Bm[N,K]^T  (bf16 in, f32 acc)
// Workgroup = 4 waves, 64x64 block tile; wave w computes rows [w*16, w*16+16)
// x 64 cols. Each wave TDM-loads its own 16-row A slice and 16-col B slice
// per 32-deep K step; s_wait_tensorcnt + barrier publishes the tiles.
// ---------------------------------------------------------------------------
__global__ __launch_bounds__(128)
void gemm_bf16_wmma(const bf16_t* __restrict__ A, const bf16_t* __restrict__ Bm,
                    float* __restrict__ Cf, bf16_t* __restrict__ Cb,
                    int M, int N, int K, int outF32) {
    __shared__ __align__(16) bf16_t Atile[2][64 * 32];
    __shared__ __align__(16) bf16_t Btile[2][64 * 32];

    const int lane = threadIdx.x & 31;
    const int wv   = threadIdx.x >> 5;        // wave id 0..3
    const int l15  = lane & 15;
    const int hi8  = (lane & 16) ? 8 : 0;     // C-layout row offset
    const int aoff = (lane & 16) ? 16 : 0;    // A-frag K base
    const int boff = (lane & 16) ? 16 : 0;    // B-frag K base
    const int m0 = blockIdx.x * 64;
    const int n0 = blockIdx.y * 64;

    const bf16_t* aSlice = A  + (long)(m0 + wv * 16) * K;  // 16 rows of A
    const bf16_t* bSlice = Bm + (long)(n0 + wv * 16) * K;  // 16 rows of Bm (= cols of B)

    unsigned aLds[2], bLds[2];
#pragma unroll
    for (int bsel = 0; bsel < 2; ++bsel) {
        aLds[bsel] = (unsigned)(unsigned long long)(void*)&Atile[bsel][wv * 16 * 32];
        bLds[bsel] = (unsigned)(unsigned long long)(void*)&Btile[bsel][wv * 16 * 32];
    }

    v8f acc0 = zero8(), acc1 = zero8(), acc2 = zero8(), acc3 = zero8();

    // prologue: stage k0 = 0 into buffer 0
    tdm_load_tile_2d(aLds[0], aSlice, (unsigned)K, 16u, 32u, 16u);
    tdm_load_tile_2d(bLds[0], bSlice, (unsigned)K, 16u, 32u, 16u);
    __builtin_amdgcn_s_wait_tensorcnt(0);
    __syncthreads();

    int buf = 0;
    for (int k0 = 0; k0 < K; k0 += 32) {
        const int nxt = buf ^ 1;
        const bool more = (k0 + 32) < K;
        if (more) {
            tdm_load_tile_2d(aLds[nxt], aSlice + k0 + 32, (unsigned)K, 16u, 32u, 16u);
            tdm_load_tile_2d(bLds[nxt], bSlice + k0 + 32, (unsigned)K, 16u, 32u, 16u);
        }

        const bf16_t* Ac = Atile[buf];
        const bf16_t* Bc = Btile[buf];
        v16bf a  = load_a_frag(Ac + (wv * 16 + l15) * 32 + aoff);
        v16bf b0 = load_b_frag(Bc + l15 * 32 + boff);
        v16bf b1 = load_b_frag(Bc + (16 + l15) * 32 + boff);
        v16bf b2 = load_b_frag(Bc + (32 + l15) * 32 + boff);
        v16bf b3 = load_b_frag(Bc + (48 + l15) * 32 + boff);
        acc0 = __builtin_amdgcn_wmma_f32_16x16x32_bf16(false, a, false, b0, (short)0, acc0, false, false);
        acc1 = __builtin_amdgcn_wmma_f32_16x16x32_bf16(false, a, false, b1, (short)0, acc1, false, false);
        acc2 = __builtin_amdgcn_wmma_f32_16x16x32_bf16(false, a, false, b2, (short)0, acc2, false, false);
        acc3 = __builtin_amdgcn_wmma_f32_16x16x32_bf16(false, a, false, b3, (short)0, acc3, false, false);

        if (more) __builtin_amdgcn_s_wait_tensorcnt(0);
        __syncthreads();   // all waves' TDM complete + all reads of buf done
        buf = nxt;
    }

#pragma unroll
    for (int r = 0; r < 8; ++r) {
        long base = (long)(m0 + wv * 16 + r + hi8) * N + n0 + l15;
        if (outF32) {
            Cf[base]      = acc0[r];
            Cf[base + 16] = acc1[r];
            Cf[base + 32] = acc2[r];
            Cf[base + 48] = acc3[r];
        } else {
            Cb[base]      = f2bf(acc0[r]);
            Cb[base + 16] = f2bf(acc1[r]);
            Cb[base + 32] = f2bf(acc2[r]);
            Cb[base + 48] = f2bf(acc3[r]);
        }
    }
}

// ---------------------------------------------------------------------------
// Kernel 3: RoPE applied in-place to q,k sections of qkv [ROWS, 6144]
// ---------------------------------------------------------------------------
__global__ void rope_kernel(bf16_t* __restrict__ qkv) {
    int idx = blockIdx.x * blockDim.x + threadIdx.x;   // ROWS * 2048 threads
    if (idx >= ROWS * 2048) return;
    int row = idx >> 11;           // 0..4095
    int c   = idx & 2047;
    int j   = c & 63;              // rotary pair index
    int sh  = c >> 6;              // (section*16 + head), 0..31 (q:0..15, k:16..31)
    int l   = row & (SEQ - 1);     // position

    long col1 = (long)row * QKVDIM + (long)sh * HDIM + j;
    long col2 = col1 + 64;

    float ang = (float)l * __expf(-0.14391160f * (float)j);   // l * 10000^(-j/64)
    float sn, cs;
    __sincosf(ang, &sn, &cs);

    float x1 = bf2f(qkv[col1]);
    float x2 = bf2f(qkv[col2]);
    qkv[col1] = f2bf(x1 * cs - x2 * sn);
    qkv[col2] = f2bf(x2 * cs + x1 * sn);
}

// ---------------------------------------------------------------------------
// Kernel 4: transpose V section of qkv into vT[(b*NH+h)*HDIM + hd][l]
// ---------------------------------------------------------------------------
__global__ void vtrans_kernel(const bf16_t* __restrict__ qkv,
                              bf16_t* __restrict__ vT) {
    long idx = (long)blockIdx.x * blockDim.x + threadIdx.x; // 2*16*128*2048
    if (idx >= (long)BATCH * NHEADS * HDIM * SEQ) return;
    int l  = (int)(idx & (SEQ - 1));
    int hd = (int)((idx >> 11) & (HDIM - 1));
    int bh = (int)(idx >> 18);                // 0..31
    int b  = bh >> 4, h = bh & 15;
    vT[idx] = qkv[(long)(b * SEQ + l) * QKVDIM + 2 * HID + h * HDIM + hd];
}

// ---------------------------------------------------------------------------
// Kernel 5: flash attention (causal, online softmax), one wave per
// (b, h, 16-query tile). Streams 32-key blocks; 16 WMMAs per block.
// ---------------------------------------------------------------------------
__global__ __launch_bounds__(32)
void flash_attn_wmma(const bf16_t* __restrict__ qkv,
                     const bf16_t* __restrict__ vT,
                     bf16_t* __restrict__ out) {
    __shared__ __align__(16) bf16_t Plds[16 * 32];

    const int lane  = threadIdx.x;
    const int l15   = lane & 15;
    const int hi8   = (lane & 16) ? 8 : 0;
    const int aoff  = (lane & 16) ? 16 : 0;
    const int boff  = (lane & 16) ? 16 : 0;
    const int qbase = blockIdx.x * 16;
    const int h     = blockIdx.y;
    const int b     = blockIdx.z;
    const float scale = 0.08838834764831845f;   // 1/sqrt(128)

    const bf16_t* qptr = qkv + (long)(b * SEQ + qbase + l15) * QKVDIM + h * HDIM + aoff;
    v16bf qf[4];
#pragma unroll
    for (int t = 0; t < 4; ++t) qf[t] = load_a_frag(qptr + 32 * t);

    const bf16_t* kbase = qkv + (long)b * SEQ * QKVDIM + HID + h * HDIM + boff;
    const bf16_t* vbase = vT + (long)(b * NHEADS + h) * HDIM * SEQ;

    v8f o[8];
#pragma unroll
    for (int t = 0; t < 8; ++t) o[t] = zero8();
    float mrow[8], lrow[8];
#pragma unroll
    for (int r = 0; r < 8; ++r) { mrow[r] = -1e30f; lrow[r] = 0.0f; }

    for (int kb = 0; kb < qbase + 16; kb += 32) {
        // ---- S = Q K^T for 16x32 key block (two 16x16 N-subtiles)
        v8f s0 = zero8(), s1 = zero8();
#pragma unroll
        for (int t = 0; t < 4; ++t) {
            v16bf k0 = load_b_frag(kbase + (long)(kb + l15)      * QKVDIM + 32 * t);
            v16bf k1 = load_b_frag(kbase + (long)(kb + 16 + l15) * QKVDIM + 32 * t);
            s0 = __builtin_amdgcn_wmma_f32_16x16x32_bf16(false, qf[t], false, k0, (short)0, s0, false, false);
            s1 = __builtin_amdgcn_wmma_f32_16x16x32_bf16(false, qf[t], false, k1, (short)0, s1, false, false);
        }
        // ---- scale + causal mask
        const bool needMask = (kb + 31) > qbase;
#pragma unroll
        for (int r = 0; r < 8; ++r) {
            int qrow = qbase + r + hi8;
            float v0 = s0[r] * scale;
            float v1 = s1[r] * scale;
            if (needMask) {
                if (kb + l15      > qrow) v0 = -1e30f;
                if (kb + 16 + l15 > qrow) v1 = -1e30f;
            }
            s0[r] = v0; s1[r] = v1;
        }
        // ---- online softmax: row max + correction + exp + row sum
        float corr[8];
#pragma unroll
        for (int r = 0; r < 8; ++r) {
            float mx = fmaxf(s0[r], s1[r]);
#pragma unroll
            for (int off = 1; off < 16; off <<= 1)
                mx = fmaxf(mx, __shfl_xor(mx, off, 16));
            float mnew = fmaxf(mrow[r], mx);
            corr[r] = __expf(mrow[r] - mnew);
            mrow[r] = mnew;
            float p0 = __expf(s0[r] - mnew);
            float p1 = __expf(s1[r] - mnew);
            s0[r] = p0; s1[r] = p1;
            float sum = p0 + p1;
#pragma unroll
            for (int off = 1; off < 16; off <<= 1)
                sum += __shfl_xor(sum, off, 16);
            lrow[r] = lrow[r] * corr[r] + sum;
        }
        // ---- rescale accumulated O
#pragma unroll
        for (int t = 0; t < 8; ++t)
#pragma unroll
            for (int r = 0; r < 8; ++r) o[t][r] *= corr[r];

        // ---- stage P (16x32) through LDS: C-layout -> A-fragment layout
        __syncthreads();
#pragma unroll
        for (int r = 0; r < 8; ++r) {
            int m = r + hi8;
            Plds[m * 32 + l15]      = f2bf(s0[r]);
            Plds[m * 32 + 16 + l15] = f2bf(s1[r]);
        }
        __syncthreads();
        v16bf pf = load_a_frag(&Plds[l15 * 32 + aoff]);

        // ---- O += P V for all 8 head-dim subtiles
#pragma unroll
        for (int t = 0; t < 8; ++t) {
            v16bf vf = load_b_frag(vbase + (long)(16 * t + l15) * SEQ + kb + boff);
            o[t] = __builtin_amdgcn_wmma_f32_16x16x32_bf16(false, pf, false, vf, (short)0, o[t], false, false);
        }
    }

    // ---- epilogue: normalize and store (bf16, [ROWS, HID] with head concat)
#pragma unroll
    for (int t = 0; t < 8; ++t) {
#pragma unroll
        for (int r = 0; r < 8; ++r) {
            int m = r + hi8;
            float val = o[t][r] / lrow[r];
            out[(long)(b * SEQ + qbase + m) * HID + h * HDIM + 16 * t + l15] = f2bf(val);
        }
    }
}

// ---------------------------------------------------------------------------
// Launch
// ---------------------------------------------------------------------------
extern "C" void kernel_launch(void* const* d_in, const int* in_sizes, int n_in,
                              void* d_out, int out_size, void* d_ws, size_t ws_size,
                              hipStream_t stream) {
    const float* x    = (const float*)d_in[0];   // (2, 2048, 2048)
    const float* Wqkv = (const float*)d_in[1];   // (6144, 2048)
    const float* Wo   = (const float*)d_in[2];   // (2048, 2048)
    float* out        = (float*)d_out;           // (2, 2048, 2048)

    char* ws = (char*)d_ws;
    bf16_t* xb    = (bf16_t*)(ws);                               // 16.78 MB
    bf16_t* wqkvb = (bf16_t*)(ws + 16777216L);                   // 25.17 MB
    bf16_t* wob   = (bf16_t*)(ws + 41943040L);                   //  8.39 MB
    bf16_t* qkvb  = (bf16_t*)(ws + 50331648L);                   // 50.33 MB
    bf16_t* vT    = (bf16_t*)(ws + 100663296L);                  // 16.78 MB
    bf16_t* ao    = (bf16_t*)(ws + 117440512L);                  // 16.78 MB
    // total: 128 MB

    // 1. convert inputs to bf16
    {
        long n4;
        n4 = (long)ROWS * HID / 4;
        cvt_f32_to_bf16<<<dim3((unsigned)((n4 + 255) / 256)), 256, 0, stream>>>(x, xb, n4);
        n4 = (long)QKVDIM * HID / 4;
        cvt_f32_to_bf16<<<dim3((unsigned)((n4 + 255) / 256)), 256, 0, stream>>>(Wqkv, wqkvb, n4);
        n4 = (long)HID * HID / 4;
        cvt_f32_to_bf16<<<dim3((unsigned)((n4 + 255) / 256)), 256, 0, stream>>>(Wo, wob, n4);
    }

    // 2. QKV projection: [4096, 6144] = xb [4096,2048] * Wqkv^T
    gemm_bf16_wmma<<<dim3(ROWS / 64, QKVDIM / 64), 128, 0, stream>>>(
        xb, wqkvb, nullptr, qkvb, ROWS, QKVDIM, HID, 0);

    // 3. RoPE on q,k
    rope_kernel<<<dim3((ROWS * 2048) / 256), 256, 0, stream>>>(qkvb);

    // 4. transpose V for contiguous B-fragment loads
    vtrans_kernel<<<dim3((unsigned)(((long)BATCH * NHEADS * HDIM * SEQ) / 256)), 256, 0, stream>>>(
        qkvb, vT);

    // 5. causal flash attention
    flash_attn_wmma<<<dim3(SEQ / 16, NHEADS, BATCH), 32, 0, stream>>>(qkvb, vT, ao);

    // 6. output projection: out [4096,2048] = ao * Wo^T (fp32 result)
    gemm_bf16_wmma<<<dim3(ROWS / 64, HID / 64), 128, 0, stream>>>(
        ao, wob, out, nullptr, ROWS, HID, HID, 1);
}
